// OccDenseDecoder_37349035606794
// MI455X (gfx1250) — matrix operand order repaired and checked by reading, not compiled, not run on hardware
//
#include <hip/hip_runtime.h>
#include <hip/hip_bf16.h>
#include <stdint.h>
#include <stddef.h>

// ---------------------------------------------------------------------------
// Types for CDNA5 WMMA (wave32): bf16 A/B (16 x bf16 per lane = 8 VGPRs),
// f32 accumulator (8 VGPRs).
// ---------------------------------------------------------------------------
typedef __bf16 bf16_t;
typedef __attribute__((ext_vector_type(16))) bf16_t       v16bf;
typedef __attribute__((ext_vector_type(8)))  float        v8f;
typedef __attribute__((ext_vector_type(4)))  unsigned int u32x4;

union FragAB { u32x4 q[2]; v16bf v; };

__device__ __forceinline__ unsigned short f2bf(float f) {
  union { float f; unsigned int u; } c; c.f = f;
  unsigned int u = c.u;
  unsigned int r = (u + 0x7FFFu + ((u >> 16) & 1u)) >> 16;  // RNE
  return (unsigned short)r;
}

// ---------------------------------------------------------------------------
// Utility fills / table build / dtype conversion
// ---------------------------------------------------------------------------
__global__ void fill_u32_kernel(unsigned int* __restrict__ p, unsigned int v, int n) {
  int i = blockIdx.x * blockDim.x + threadIdx.x;
  if (i < n) p[i] = v;
}

__global__ void build_tbl_kernel(const int* __restrict__ coords, int n,
                                 int D1, int D2, int D3, int* __restrict__ tbl) {
  int i = blockIdx.x * blockDim.x + threadIdx.x;
  if (i >= n) return;
  int c0 = coords[i * 4 + 0], c1 = coords[i * 4 + 1];
  int c2 = coords[i * 4 + 2], c3 = coords[i * 4 + 3];
  tbl[((c0 * D1 + c1) * D2 + c2) * D3 + c3] = i;
}

// feats (n x 48 f32) -> bf16, row stride 64 (pad channels 48..63 with 0)
__global__ void cvt_feats_kernel(const float* __restrict__ feats, int n,
                                 unsigned short* __restrict__ out) {
  int idx = blockIdx.x * blockDim.x + threadIdx.x;
  if (idx >= n * 64) return;
  int i = idx >> 6, c = idx & 63;
  out[idx] = (c < 48) ? f2bf(feats[(size_t)i * 48 + c]) : (unsigned short)0;
}

// w (81 x 48(ci) x 48(co) f32) -> wbt[k][co][ci_pad64] bf16 (B^T layout)
__global__ void cvt_w_kernel(const float* __restrict__ w, unsigned short* __restrict__ wbt) {
  int idx = blockIdx.x * blockDim.x + threadIdx.x;
  if (idx >= 81 * 48 * 64) return;
  int k = idx / 3072, r = idx % 3072;
  int co = r >> 6, ci = r & 63;
  wbt[idx] = (ci < 48) ? f2bf(w[((size_t)k * 48 + ci) * 48 + co]) : (unsigned short)0;
}

// ---------------------------------------------------------------------------
// Sparse submanifold conv: acc[i,co] = sum_k sum_ci gather(feats,k)[i,ci]*W[k][ci][co]
// Block = 384 threads (12 waves) = 4 M-tiles x 3 N-tiles over a 64-row tile.
// B tile (48x64 bf16) staged once per offset and reused by all 4 M-tiles,
// cutting L2 weight traffic 4x vs a 16-row tile. K padded 48->64 -> two
// v_wmma_f32_16x16x32_bf16 per wave per offset.
// ---------------------------------------------------------------------------
#define ALDS_STRIDE 72   // ushorts per A row (144 B, 16B aligned, bank-skewed)
#define BLDS_STRIDE 72
#define MROWS       64   // query rows per block
#define NTHREADS    384

__global__ void __launch_bounds__(NTHREADS)
sparse_conv_wmma_kernel(const int* __restrict__ qlat, int m,
                        int sx, int sy, int sz, int st,
                        const int* __restrict__ tbl,
                        int D0, int D1, int D2, int D3,
                        const unsigned short* __restrict__ featsb,   // rows of 64 bf16
                        const unsigned short* __restrict__ wbt,      // 81*48*64 bf16
                        float* __restrict__ xout) {                  // m x 48 f32
  __shared__ unsigned short Ash[MROWS * ALDS_STRIDE];  // 9216 B
  __shared__ unsigned short Bsh[48 * BLDS_STRIDE];     // 6912 B
  __shared__ int idxsh[MROWS];

  const int tid  = threadIdx.x;
  const int lane = tid & 31;
  const int wave = tid >> 5;       // 0..11
  const int mt   = wave >> 2;      // wrong split? -> use wave/3 mapping below
  (void)mt;
  const int mtile = wave / 3;      // 0..3
  const int ntile = wave % 3;      // 0..2
  const int tile  = blockIdx.x;

  v8f acc = {0.f, 0.f, 0.f, 0.f, 0.f, 0.f, 0.f, 0.f};

  for (int k = 0; k < 81; ++k) {
    const int o0 = k / 27 - 1;
    const int o1 = (k / 9) % 3 - 1;
    const int o2 = (k / 3) % 3 - 1;
    const int o3 = k % 3 - 1;

    // ---- stage B tile (48 x 64 bf16 = 6 KB): 384 16B chunks, one per thread
    {
      const unsigned short* wsrc = wbt + (size_t)k * 3072;
      int co = tid >> 3, ch = tid & 7;
      *(u32x4*)(&Bsh[co * BLDS_STRIDE + ch * 8]) =
          *(const u32x4*)(wsrc + co * 64 + ch * 8);
    }

    // ---- neighbor lookup for the 64 query rows
    if (tid < MROWS) {
      int row = tile * MROWS + tid;
      if (row >= m) row = m - 1;
      int q0 = qlat[row * 4 + 0] * sx + o0;
      int q1 = qlat[row * 4 + 1] * sy + o1;
      int q2 = qlat[row * 4 + 2] * sz + o2;
      int q3 = qlat[row * 4 + 3] * st + o3;
      int idx = -1;
      if ((unsigned)q0 < (unsigned)D0 && (unsigned)q1 < (unsigned)D1 &&
          (unsigned)q2 < (unsigned)D2 && (unsigned)q3 < (unsigned)D3) {
        idx = tbl[((q0 * D1 + q1) * D2 + q2) * D3 + q3];
      }
      idxsh[tid] = idx;
    }
    __syncthreads();

    // ---- stage A tile (64 rows x 64 bf16): 512 16B chunks over 384 threads
    for (int id = tid; id < MROWS * 8; id += NTHREADS) {
      int r = id >> 3, ch = id & 7;
      int idx = idxsh[r];
      u32x4 val = {0u, 0u, 0u, 0u};
      if (idx >= 0) val = *(const u32x4*)(featsb + (size_t)idx * 64 + ch * 8);
      *(u32x4*)(&Ash[r * ALDS_STRIDE + ch * 8]) = val;
    }
    __syncthreads();

    // prefetch next offset's weight tile into the cache hierarchy
    if (k + 1 < 81 && tid < 8)
      __builtin_prefetch(wbt + (size_t)(k + 1) * 3072 + tid * 384, 0, 1);

    // ---- 2 x v_wmma_f32_16x16x32_bf16 per wave (K = 64 padded)
    {
      const int hlf  = lane >> 4;        // lane half selects K sub-range
      const int mrow = mtile * 16 + (lane & 15);  // A row within block tile
      const int ncol = lane & 15;                 // B column within N tile
#pragma unroll
      for (int s = 0; s < 2; ++s) {
        FragAB a, b;
        // A layout: j0..7 -> K = 32s + 8h + (0..7); j8..15 -> K = 32s + 16 + 8h + (0..7)
        a.q[0] = *(const u32x4*)(&Ash[mrow * ALDS_STRIDE + s * 32 + 8 * hlf]);
        a.q[1] = *(const u32x4*)(&Ash[mrow * ALDS_STRIDE + s * 32 + 16 + 8 * hlf]);
        // B layout: j0..7 -> K = 32s + 8h + (0..7); j8..15 -> K = 32s + 16 + 8h + (0..7), column = lane&15
        const unsigned short* bp =
            &Bsh[(ntile * 16 + ncol) * BLDS_STRIDE + s * 32 + 8 * hlf];
        b.q[0] = *(const u32x4*)(bp);
        b.q[1] = *(const u32x4*)(bp + 16);
        acc = __builtin_amdgcn_wmma_f32_16x16x32_bf16(
            false, a.v, false, b.v, (short)0, acc, false, false);
      }
    }
    __syncthreads();
  }

  // ---- write D tile: lane holds D[m = r + 8*(lane>>4)][n = lane&15], r = 0..7
  const int hlf  = lane >> 4;
  const int ncol = lane & 15;
#pragma unroll
  for (int r = 0; r < 8; ++r) {
    int row = tile * MROWS + mtile * 16 + r + 8 * hlf;
    if (row < m) xout[(size_t)row * 48 + ntile * 16 + ncol] = acc[r];
  }
}

// ---------------------------------------------------------------------------
// Row batch-norm: reduce (one block per channel) then apply
// ---------------------------------------------------------------------------
__global__ void bn_reduce_kernel(const float* __restrict__ x, int m,
                                 float* __restrict__ stats) {  // [48]=sum, [48..95]=sumsq
  int c = blockIdx.x;
  __shared__ float ssum[256], ssq[256];
  float s = 0.f, q = 0.f;
  for (int i = threadIdx.x; i < m; i += 256) {
    float v = x[(size_t)i * 48 + c];
    s += v; q += v * v;
  }
  ssum[threadIdx.x] = s; ssq[threadIdx.x] = q;
  __syncthreads();
  for (int off = 128; off > 0; off >>= 1) {
    if (threadIdx.x < off) {
      ssum[threadIdx.x] += ssum[threadIdx.x + off];
      ssq[threadIdx.x]  += ssq[threadIdx.x + off];
    }
    __syncthreads();
  }
  if (threadIdx.x == 0) { stats[c] = ssum[0]; stats[48 + c] = ssq[0]; }
}

__global__ void bn_apply_kernel(const float* __restrict__ x, int m,
                                const float* __restrict__ stats,
                                const float* __restrict__ g, const float* __restrict__ b,
                                float* __restrict__ of32,            // may be null
                                unsigned short* __restrict__ obf) {  // may be null, stride 64
  int idx = blockIdx.x * blockDim.x + threadIdx.x;
  if (idx >= m * 64) return;
  int i = idx >> 6, c = idx & 63;
  if (c < 48) {
    float mean = stats[c] / (float)m;
    float var  = stats[48 + c] / (float)m - mean * mean;
    float v = g[c] * (x[(size_t)i * 48 + c] - mean) * rsqrtf(var + 1e-5f) + b[c];
    if (of32) of32[(size_t)i * 48 + c] = v;
    if (obf)  obf[idx] = f2bf(v);
  } else if (obf) {
    obf[idx] = 0;
  }
}

// ---------------------------------------------------------------------------
// Scatter-add sparse rows into dense (C, 96, 96) image
// ---------------------------------------------------------------------------
__global__ void scatter_dense_kernel(const float* __restrict__ x2,
                                     const int* __restrict__ lat, int m,
                                     float* __restrict__ h0) {
  int idx = blockIdx.x * blockDim.x + threadIdx.x;
  if (idx >= m * 48) return;
  int i = idx / 48, c = idx % 48;
  int px = lat[i * 4 + 0], py = lat[i * 4 + 1];
  atomicAdd(&h0[(size_t)c * 9216 + px * 96 + py], x2[idx]);
}

// ---------------------------------------------------------------------------
// Dense 3x3 conv (SAME), NCHW 1x48x96x96. Tiny FLOP share -> direct VALU.
// ---------------------------------------------------------------------------
__global__ void conv3x3_kernel(const float* __restrict__ in,
                               const float* __restrict__ w,     // [48][48][3][3]
                               const float* __restrict__ bias,  // [48]
                               float* __restrict__ out) {
  int idx = blockIdx.x * blockDim.x + threadIdx.x;
  if (idx >= 48 * 9216) return;
  int co = idx / 9216, p = idx % 9216;
  int px = p / 96, py = p % 96;
  float acc = bias[co];
  for (int ci = 0; ci < 48; ++ci) {
    const float* ip = in + (size_t)ci * 9216;
    const float* wp = w + ((size_t)co * 48 + ci) * 9;
#pragma unroll
    for (int ky = 0; ky < 3; ++ky) {
      int xx = px + ky - 1;
      if ((unsigned)xx >= 96u) continue;
#pragma unroll
      for (int kx = 0; kx < 3; ++kx) {
        int yy = py + kx - 1;
        if ((unsigned)yy < 96u) acc += ip[xx * 96 + yy] * wp[ky * 3 + kx];
      }
    }
  }
  out[idx] = acc;
}

__global__ void bn2_reduce_kernel(const float* __restrict__ x, float* __restrict__ stats) {
  int c = blockIdx.x;
  __shared__ float ssum[256], ssq[256];
  float s = 0.f, q = 0.f;
  for (int i = threadIdx.x; i < 9216; i += 256) {
    float v = x[(size_t)c * 9216 + i];
    s += v; q += v * v;
  }
  ssum[threadIdx.x] = s; ssq[threadIdx.x] = q;
  __syncthreads();
  for (int off = 128; off > 0; off >>= 1) {
    if (threadIdx.x < off) {
      ssum[threadIdx.x] += ssum[threadIdx.x + off];
      ssq[threadIdx.x]  += ssq[threadIdx.x + off];
    }
    __syncthreads();
  }
  if (threadIdx.x == 0) { stats[c] = ssum[0]; stats[48 + c] = ssq[0]; }
}

__global__ void bn2_apply_relu_kernel(float* __restrict__ x,
                                      const float* __restrict__ stats,
                                      const float* __restrict__ g,
                                      const float* __restrict__ b) {
  int idx = blockIdx.x * blockDim.x + threadIdx.x;
  if (idx >= 48 * 9216) return;
  int c = idx / 9216;
  float mean = stats[c] * (1.f / 9216.f);
  float var  = stats[48 + c] * (1.f / 9216.f) - mean * mean;
  float v = g[c] * (x[idx] - mean) * rsqrtf(var + 1e-5f) + b[c];
  x[idx] = fmaxf(v, 0.f);
}

// out[(t*12+z)*96*96 + y*96 + x] = relu(h[(z*4+t)*9216 + x*96 + y])
__global__ void final_out_kernel(const float* __restrict__ h, float* __restrict__ out) {
  int idx = blockIdx.x * blockDim.x + threadIdx.x;
  if (idx >= 4 * 12 * 96 * 96) return;
  int x = idx % 96;
  int y = (idx / 96) % 96;
  int z = (idx / 9216) % 12;
  int t = idx / (9216 * 12);
  float v = h[(size_t)(z * 4 + t) * 9216 + x * 96 + y];
  out[idx] = fmaxf(v, 0.f);
}

// ---------------------------------------------------------------------------
// Host launcher (graph-capture safe: kernel launches only)
// ---------------------------------------------------------------------------
extern "C" void kernel_launch(void* const* d_in, const int* in_sizes, int n_in,
                              void* d_out, int out_size, void* d_ws, size_t ws_size,
                              hipStream_t stream) {
  const int*   coords  = (const int*)d_in[0];
  const int*   out_lat = (const int*)d_in[1];
  const float* feats   = (const float*)d_in[2];
  const float* w0      = (const float*)d_in[3];
  const float* w1      = (const float*)d_in[4];
  const float* bn0_g   = (const float*)d_in[5];
  const float* bn0_b   = (const float*)d_in[6];
  const float* bn1_g   = (const float*)d_in[7];
  const float* bn1_b   = (const float*)d_in[8];
  const float* conv_w  = (const float*)d_in[9];   // (5,48,48,3,3)
  const float* conv_b  = (const float*)d_in[10];  // (5,48)
  const float* dbn_g   = (const float*)d_in[11];  // (4,48)
  const float* dbn_b   = (const float*)d_in[12];  // (4,48)
  float* out = (float*)d_out;

  const int n = in_sizes[2] / 48;  // feats rows
  const int m = in_sizes[1] / 4;   // out_lat rows

  const int GRID_N = 192 * 192 * 24 * 4;  // 3,538,944
  const int LAT_N  = 96 * 96 * 12 * 4;    // 442,368
  const int IMG    = 48 * 96 * 96;        // 442,368

  // bump allocator over d_ws
  char* wsb = (char*)d_ws;
  size_t off = 0;
  auto alloc = [&](size_t bytes) -> void* {
    off = (off + 255) & ~(size_t)255;
    void* p = wsb + off;
    off += bytes;
    return p;
  };

  int* tbl_in  = (int*)alloc((size_t)GRID_N * 4);
  int* tbl_lat = (int*)alloc((size_t)LAT_N * 4);
  unsigned short* featsb = (unsigned short*)alloc((size_t)n * 64 * 2);
  unsigned short* w0bt   = (unsigned short*)alloc((size_t)81 * 3072 * 2);
  unsigned short* w1bt   = (unsigned short*)alloc((size_t)81 * 3072 * 2);
  float* x1 = (float*)alloc((size_t)m * 48 * 4);
  unsigned short* x1b = (unsigned short*)alloc((size_t)m * 64 * 2);
  float* x2 = (float*)alloc((size_t)m * 48 * 4);
  float* stats = (float*)alloc(1024 * 4);  // st0@0, st1@96, dense bn i @ 192+96i
  float* h0 = (float*)alloc((size_t)IMG * 4);
  float* hA = (float*)alloc((size_t)IMG * 4);
  float* hB = (float*)alloc((size_t)IMG * 4);

  float* st0 = stats;
  float* st1 = stats + 96;

  const int T = 256;

  // ---- init tables / accumulators (every call: deterministic)
  fill_u32_kernel<<<(GRID_N + T - 1) / T, T, 0, stream>>>((unsigned int*)tbl_in, 0xFFFFFFFFu, GRID_N);
  fill_u32_kernel<<<(LAT_N + T - 1) / T, T, 0, stream>>>((unsigned int*)tbl_lat, 0xFFFFFFFFu, LAT_N);
  fill_u32_kernel<<<(1024 + T - 1) / T, T, 0, stream>>>((unsigned int*)stats, 0u, 1024);
  fill_u32_kernel<<<(IMG + T - 1) / T, T, 0, stream>>>((unsigned int*)h0, 0u, IMG);

  build_tbl_kernel<<<(n + T - 1) / T, T, 0, stream>>>(coords, n, 192, 24, 4, tbl_in);
  build_tbl_kernel<<<(m + T - 1) / T, T, 0, stream>>>(out_lat, m, 96, 12, 4, tbl_lat);

  cvt_feats_kernel<<<((size_t)n * 64 + T - 1) / T, T, 0, stream>>>(feats, n, featsb);
  cvt_w_kernel<<<(81 * 3072 + T - 1) / T, T, 0, stream>>>(w0, w0bt);
  cvt_w_kernel<<<(81 * 3072 + T - 1) / T, T, 0, stream>>>(w1, w1bt);

  // ---- sparse conv 1 (queries = out_lat * stride, table over GRID)
  dim3 scg((m + MROWS - 1) / MROWS);
  sparse_conv_wmma_kernel<<<scg, NTHREADS, 0, stream>>>(
      out_lat, m, 2, 2, 2, 1, tbl_in, 192, 192, 24, 4, featsb, w0bt, x1);
  bn_reduce_kernel<<<48, 256, 0, stream>>>(x1, m, st0);
  bn_apply_kernel<<<((size_t)m * 64 + T - 1) / T, T, 0, stream>>>(
      x1, m, st0, bn0_g, bn0_b, nullptr, x1b);

  // ---- sparse conv 2 (queries = out_lat, table over LAT)
  sparse_conv_wmma_kernel<<<scg, NTHREADS, 0, stream>>>(
      out_lat, m, 1, 1, 1, 1, tbl_lat, 96, 96, 12, 4, x1b, w1bt, x2);
  bn_reduce_kernel<<<48, 256, 0, stream>>>(x2, m, st1);
  bn_apply_kernel<<<((size_t)m * 64 + T - 1) / T, T, 0, stream>>>(
      x2, m, st1, bn1_g, bn1_b, x2, nullptr);  // in-place f32

  // ---- densify
  scatter_dense_kernel<<<((size_t)m * 48 + T - 1) / T, T, 0, stream>>>(x2, out_lat, m, h0);

  // ---- dense conv stack
  const int cb = (IMG + T - 1) / T;
  float* hin = h0;
  float* ping[2] = {hA, hB};
  for (int i = 0; i < 4; ++i) {
    float* hnext = ping[i & 1];
    conv3x3_kernel<<<cb, T, 0, stream>>>(hin, conv_w + (size_t)i * 48 * 48 * 9,
                                         conv_b + i * 48, hnext);
    float* sti = stats + 192 + 96 * i;
    bn2_reduce_kernel<<<48, 256, 0, stream>>>(hnext, sti);
    bn2_apply_relu_kernel<<<cb, T, 0, stream>>>(hnext, sti, dbn_g + i * 48, dbn_b + i * 48);
    hin = hnext;
  }
  // layer 4: conv + bias only (hin == hB after 4 layers -> write to hA)
  conv3x3_kernel<<<cb, T, 0, stream>>>(hin, conv_w + (size_t)4 * 48 * 48 * 9,
                                       conv_b + 4 * 48, hA);

  // ---- relu + permute to (1, TS, ZH, Yo, Xo)
  final_out_kernel<<<(4 * 12 * 96 * 96 + T - 1) / T, T, 0, stream>>>(hA, out);
}
